// GATv2Backbone_48189533061603
// MI455X (gfx1250) — matrix operand order repaired and checked by reading, not compiled, not run on hardware
//
#include <hip/hip_runtime.h>
#include <math.h>

#define N_NODES 20000
#define N_EDGES 320000
#define E_TOT   (N_EDGES + N_NODES)   // self-loops appended
#define HEADS   8
#define DIN     256                   // K of every GEMM
#define KSTEPS  (DIN / 32)

typedef __attribute__((ext_vector_type(16))) __bf16 bf16x16;
typedef __attribute__((ext_vector_type(8)))  __bf16 bf16x8;
typedef __attribute__((ext_vector_type(8)))  float  floatx8;

// ---------------------------------------------------------------------------
// One-time fp32 -> bf16 hi/lo split (row-major, used for the A operand).
// a ~= hi + lo with hi = bf16(a), lo = bf16(a - hi)  -> ~2^-18 relative error.
// ---------------------------------------------------------------------------
__global__ void k_split_bf16(const float* __restrict__ x, __bf16* __restrict__ hi,
                             __bf16* __restrict__ lo, int n) {
  int i = blockIdx.x * blockDim.x + threadIdx.x;
  if (i >= n) return;
  float v = x[i];
  __bf16 h = (__bf16)v;
  hi[i] = h;
  lo[i] = (__bf16)(v - (float)h);
}

// ---------------------------------------------------------------------------
// Pack W[K,N] (row-major fp32) into WMMA B-fragment order, split hi/lo:
//   packed[((ntile*KSTEPS + ks)*32 + lane)*16 + i]
//     = bf16( W[(ks*32 + (lane>>4)*8 + (i<8 ? i : i+8))*N + ntile*16 + (lane&15)] )
// so each lane's 16-element fragment is one contiguous 32B load in the GEMM.
// ---------------------------------------------------------------------------
__global__ void k_pack_w(const float* __restrict__ W, __bf16* __restrict__ phi,
                         __bf16* __restrict__ plo, int N) {
  int idx = blockIdx.x * blockDim.x + threadIdx.x;
  const int ntiles = N >> 4;
  const int total = ntiles * KSTEPS * 32;
  if (idx >= total) return;
  const int lane = idx & 31;
  const int ks   = (idx >> 5) % KSTEPS;
  const int nt   = (idx >> 5) / KSTEPS;
  const int col  = (nt << 4) + (lane & 15);
  const int kb   = (ks << 5) + ((lane >> 4) << 4);
  const size_t base = (size_t)idx * 16;
  for (int i = 0; i < 16; ++i) {
    float v = W[(size_t)(kb - ((lane >> 4) << 3) + (i < 8 ? i : i + 8)) * N + col];
    __bf16 h = (__bf16)v;
    phi[base + i] = h;
    plo[base + i] = (__bf16)(v - (float)h);
  }
}

// ---------------------------------------------------------------------------
// GEMM: C[M,N] = A[M,256] @ W[256,N] + b[N], bf16x3 split products with fp32
// accumulate via v_wmma_f32_16x16x32_bf16. One wave32 per 16x16 tile.
// A hi/lo are plain row-major bf16; B hi/lo are pre-packed fragment-major.
// Fully unrolled K loop: 24 static WMMAs, loads hoistable across hazards.
// ---------------------------------------------------------------------------
__global__ __launch_bounds__(32)
void gemm_bias_wmma(const __bf16* __restrict__ Ahi, const __bf16* __restrict__ Alo,
                    const __bf16* __restrict__ Bhi, const __bf16* __restrict__ Blo,
                    const float* __restrict__ bias, float* __restrict__ C, int N) {
  const int lane = threadIdx.x;
  const int half = lane >> 4;          // 0: lanes 0-15, 1: lanes 16-31
  const int l    = lane & 15;
  const int m0   = blockIdx.x << 4;
  const int nt   = blockIdx.y;
  const int kbA  = half << 3;          // A K-suboffset: 0 or 8

  const __bf16* ah = Ahi + (size_t)(m0 + l) * DIN + kbA;
  const __bf16* al = Alo + (size_t)(m0 + l) * DIN + kbA;
  const __bf16* bh = Bhi + ((size_t)nt * KSTEPS * 32 + lane) * 16;
  const __bf16* bl = Blo + ((size_t)nt * KSTEPS * 32 + lane) * 16;

  floatx8 acc = {};
#pragma unroll
  for (int ks = 0; ks < KSTEPS; ++ks) {
    // A 16x32 fragment (ISA 7.12.2): elems 0-7 at K=kbA+i, elems 8-15 at K=kbA+16+i
    bf16x8 a0 = *(const bf16x8*)(ah);
    bf16x8 a1 = *(const bf16x8*)(ah + 16);
    bf16x8 c0 = *(const bf16x8*)(al);
    bf16x8 c1 = *(const bf16x8*)(al + 16);
    bf16x16 ahi16 = __builtin_shufflevector(a0, a1, 0,1,2,3,4,5,6,7,8,9,10,11,12,13,14,15);
    bf16x16 alo16 = __builtin_shufflevector(c0, c1, 0,1,2,3,4,5,6,7,8,9,10,11,12,13,14,15);
    // B 32x16 fragment: contiguous 32B per lane (pre-packed)
    bf16x16 bhi16 = *(const bf16x16*)(bh);
    bf16x16 blo16 = *(const bf16x16*)(bl);

    if (ks + 1 < KSTEPS) {             // gfx1250 global_prefetch_b8
      __builtin_prefetch(ah + 32, 0, 3);
      __builtin_prefetch(bh + 512, 0, 3);
    }

    // bf16x3: ab ~= ahi*bhi + ahi*blo + alo*bhi (fp32 accumulate)
    acc = __builtin_amdgcn_wmma_f32_16x16x32_bf16(false, ahi16, false, bhi16, (short)0, acc, false, false);
    acc = __builtin_amdgcn_wmma_f32_16x16x32_bf16(false, ahi16, false, blo16, (short)0, acc, false, false);
    acc = __builtin_amdgcn_wmma_f32_16x16x32_bf16(false, alo16, false, bhi16, (short)0, acc, false, false);

    ah += 32; al += 32; bh += 512; bl += 512;
  }

  // C/D layout: VGPR r -> row (m0 + r + 8*half), col = nt*16 + lane%16
  const int col = (nt << 4) + l;
  const float bv = bias[col];
  float* crow = C + (size_t)(m0 + (half << 3)) * N + col;
#pragma unroll
  for (int r = 0; r < 8; ++r)
    crow[(size_t)r * N] = acc[r] + bv;
}

// ---------------------------------------------------------------------------
// Edge phase (memory-bound; roofline-dominant part of the workload)
// ---------------------------------------------------------------------------
__device__ __forceinline__ void edge_nodes(const int* __restrict__ ei, int e,
                                           int& s, int& t) {
  if (e < N_EDGES) { s = ei[e]; t = ei[N_EDGES + e]; }   // edge_index[0][e], [1][e]
  else             { s = t = e - N_EDGES; }               // appended self-loop
}

__device__ __forceinline__ void atomic_max_f32(float* addr, float val) {
  if (val >= 0.0f) atomicMax((int*)addr, __float_as_int(val));
  else             atomicMin((unsigned int*)addr, __float_as_uint(val));
}

__global__ void k_init_softmax(float* __restrict__ emax, float* __restrict__ den, int n) {
  int i = blockIdx.x * blockDim.x + threadIdx.x;
  if (i < n) { emax[i] = -3.0e38f; den[i] = 0.0f; }
}

__global__ void k_zero(float* __restrict__ p, int n) {
  int i = blockIdx.x * blockDim.x + threadIdx.x;
  if (i < n) p[i] = 0.0f;
}

__global__ void k_edge_logits(const float* __restrict__ xl, const float* __restrict__ xr,
                              const float* __restrict__ att, const int* __restrict__ ei,
                              int D, float* __restrict__ elog, float* __restrict__ emax) {
  int idx = blockIdx.x * blockDim.x + threadIdx.x;
  if (idx >= E_TOT * HEADS) return;
  int e = idx >> 3, h = idx & 7;
  int s, t; edge_nodes(ei, e, s, t);
  int HD = HEADS * D;
  const float* pl = xl + (size_t)s * HD + h * D;
  const float* pr = xr + (size_t)t * HD + h * D;
  const float* pa = att + h * D;
  float sum = 0.0f;
  for (int d = 0; d < D; ++d) {
    float m = pl[d] + pr[d];
    m = (m >= 0.0f) ? m : 0.2f * m;                      // negative_slope = 0.2
    sum += m * pa[d];
  }
  elog[idx] = sum;
  atomic_max_f32(&emax[t * HEADS + h], sum);
}

__global__ void k_edge_exp(float* __restrict__ elog, const float* __restrict__ emax,
                           float* __restrict__ den, const int* __restrict__ ei) {
  int idx = blockIdx.x * blockDim.x + threadIdx.x;
  if (idx >= E_TOT * HEADS) return;
  int e = idx >> 3, h = idx & 7;
  int s, t; edge_nodes(ei, e, s, t);
  float p = expf(elog[idx] - emax[t * HEADS + h]);
  elog[idx] = p;
  atomicAdd(&den[t * HEADS + h], p);
}

__global__ void k_edge_scatter(const float* __restrict__ elog, const float* __restrict__ den,
                               const float* __restrict__ xl, const int* __restrict__ ei,
                               int D, float* __restrict__ acc) {
  int idx = blockIdx.x * blockDim.x + threadIdx.x;
  if (idx >= E_TOT * HEADS) return;
  int e = idx >> 3, h = idx & 7;
  int s, t; edge_nodes(ei, e, s, t);
  float alpha = elog[idx] / den[t * HEADS + h];
  int HD = HEADS * D;
  const float* pl = xl + (size_t)s * HD + h * D;
  float*       pd = acc + (size_t)t * HD + h * D;
  for (int d = 0; d < D; ++d)
    atomicAdd(&pd[d], alpha * pl[d]);
}

__global__ void k_elu_bias(const float* __restrict__ acc, const float* __restrict__ bias,
                           const float* __restrict__ resid, float* __restrict__ out,
                           int HD, int total) {
  int i = blockIdx.x * blockDim.x + threadIdx.x;
  if (i >= total) return;
  float v = acc[i] + bias[i % HD];
  v = (v > 0.0f) ? v : expm1f(v);                        // jax.nn.elu, alpha=1
  if (resid) v += resid[i];
  out[i] = v;
}

__global__ void k_mean_bias(const float* __restrict__ acc, const float* __restrict__ bias,
                            float* __restrict__ out, int D) {
  int i = blockIdx.x * blockDim.x + threadIdx.x;
  if (i >= N_NODES * D) return;
  int n = i / D, d = i % D;
  const float* p = acc + (size_t)n * HEADS * D + d;
  float s = 0.0f;
#pragma unroll
  for (int h = 0; h < HEADS; ++h) s += p[h * D];
  out[i] = s * (1.0f / HEADS) + bias[d];
}

// ---------------------------------------------------------------------------
extern "C" void kernel_launch(void* const* d_in, const int* in_sizes, int n_in,
                              void* d_out, int out_size, void* d_ws, size_t ws_size,
                              hipStream_t stream) {
  (void)in_sizes; (void)n_in; (void)out_size; (void)ws_size;
  const float* x  = (const float*)d_in[0];
  const int*   ei = (const int*)d_in[1];
  const float* Wl[3]  = {(const float*)d_in[2],  (const float*)d_in[8],  (const float*)d_in[14]};
  const float* bl[3]  = {(const float*)d_in[3],  (const float*)d_in[9],  (const float*)d_in[15]};
  const float* Wr[3]  = {(const float*)d_in[4],  (const float*)d_in[10], (const float*)d_in[16]};
  const float* br[3]  = {(const float*)d_in[5],  (const float*)d_in[11], (const float*)d_in[17]};
  const float* att[3] = {(const float*)d_in[6],  (const float*)d_in[12], (const float*)d_in[18]};
  const float* bias[3]= {(const float*)d_in[7],  (const float*)d_in[13], (const float*)d_in[19]};

  // workspace carve-up, 256B-aligned chunks (~198 MB total)
  char*  base = (char*)d_ws;
  size_t off  = 0;
  auto alloc = [&](size_t bytes) -> void* {
    void* p = base + off;
    off = (off + bytes + 255) & ~(size_t)255;
    return p;
  };
  float*  f_xl   = (float*)alloc((size_t)N_NODES * 512 * 4);
  float*  f_xr   = (float*)alloc((size_t)N_NODES * 512 * 4);
  float*  f_acc  = (float*)alloc((size_t)N_NODES * 512 * 4);
  float*  f_hin  = (float*)alloc((size_t)N_NODES * 256 * 4);
  float*  f_h2   = (float*)alloc((size_t)N_NODES * 256 * 4);
  float*  f_e    = (float*)alloc((size_t)E_TOT * HEADS * 4);
  float*  f_emax = (float*)alloc((size_t)N_NODES * HEADS * 4);
  float*  f_den  = (float*)alloc((size_t)N_NODES * HEADS * 4);
  __bf16* a_hi   = (__bf16*)alloc((size_t)N_NODES * DIN * 2);
  __bf16* a_lo   = (__bf16*)alloc((size_t)N_NODES * DIN * 2);
  __bf16* wl_hi  = (__bf16*)alloc((size_t)DIN * 512 * 2);
  __bf16* wl_lo  = (__bf16*)alloc((size_t)DIN * 512 * 2);
  __bf16* wr_hi  = (__bf16*)alloc((size_t)DIN * 512 * 2);
  __bf16* wr_lo  = (__bf16*)alloc((size_t)DIN * 512 * 2);

  const int EH = E_TOT * HEADS;
  const int TB = 256;
  auto cdiv = [](int a, int b) { return (a + b - 1) / b; };

  const int    Ds[3]       = {32, 32, 64};
  const float* layer_in[3] = {x, f_hin, f_h2};

  for (int L = 0; L < 3; ++L) {
    const int D  = Ds[L];
    const int HD = HEADS * D;

    // one-time operand preparation (reused 1250x by the M-tiles)
    k_split_bf16<<<cdiv(N_NODES * DIN, TB), TB, 0, stream>>>(layer_in[L], a_hi, a_lo, N_NODES * DIN);
    const int packThreads = (HD / 16) * KSTEPS * 32;
    k_pack_w<<<cdiv(packThreads, TB), TB, 0, stream>>>(Wl[L], wl_hi, wl_lo, HD);
    k_pack_w<<<cdiv(packThreads, TB), TB, 0, stream>>>(Wr[L], wr_hi, wr_lo, HD);

    dim3 ggrid(N_NODES / 16, HD / 16);
    gemm_bias_wmma<<<ggrid, 32, 0, stream>>>(a_hi, a_lo, wl_hi, wl_lo, bl[L], f_xl, HD);
    gemm_bias_wmma<<<ggrid, 32, 0, stream>>>(a_hi, a_lo, wr_hi, wr_lo, br[L], f_xr, HD);

    k_init_softmax<<<cdiv(N_NODES * HEADS, TB), TB, 0, stream>>>(f_emax, f_den, N_NODES * HEADS);
    k_zero<<<cdiv(N_NODES * HD, TB), TB, 0, stream>>>(f_acc, N_NODES * HD);
    k_edge_logits<<<cdiv(EH, TB), TB, 0, stream>>>(f_xl, f_xr, att[L], ei, D, f_e, f_emax);
    k_edge_exp<<<cdiv(EH, TB), TB, 0, stream>>>(f_e, f_emax, f_den, ei);
    k_edge_scatter<<<cdiv(EH, TB), TB, 0, stream>>>(f_e, f_den, f_xl, ei, D, f_acc);

    if (L == 0) {        // h_in1 = elu(out0 + bias0)
      k_elu_bias<<<cdiv(N_NODES * HD, TB), TB, 0, stream>>>(f_acc, bias[0], nullptr, f_hin, HD, N_NODES * HD);
    } else if (L == 1) { // h_in2 = elu(out1 + bias1) + h_in1
      k_elu_bias<<<cdiv(N_NODES * HD, TB), TB, 0, stream>>>(f_acc, bias[1], f_hin, f_h2, HD, N_NODES * HD);
    } else {             // out = mean_heads(out2) + bias2
      k_mean_bias<<<cdiv(N_NODES * Ds[2], TB), TB, 0, stream>>>(f_acc, bias[2], (float*)d_out, Ds[2]);
    }
  }
}
